// TransformerModel_4733053960628
// MI455X (gfx1250) — compile-verified, matrix operand order checked by reading, not run on hardware
//
#include <hip/hip_runtime.h>
#include <cstddef>
#include <cstdint>

// ---------------------------------------------------------------------------
// MI455X (gfx1250) implementation: all dense math routed through
// v_wmma_f32_16x16x32_bf16. Both GEMM operands stored K-contiguous (A [M,K],
// B pre-transposed to [N,K]) so LDS fragment reads are two b128 loads per
// 16x32 fragment, matching the CDNA5 WMMA 16-bit VGPR layout.
// Global->LDS staging uses CDNA5 async copies (global_load_async_to_lds_b128,
// ASYNCcnt) in a 3-deep LDS ring: two tiles in flight behind the WMMAs,
// s_wait_asynccnt 0x4 retires the older stage (async ops complete in order).
// 128x128 block, 8 waves, 32x64 per wave (2x4 WMMA tiles) -> 12 ds_load_b128
// per 8 WMMAs.
// ---------------------------------------------------------------------------

typedef unsigned short u16;
typedef unsigned int   u32;
typedef float  v8f   __attribute__((ext_vector_type(8)));
typedef __bf16 v16bf __attribute__((ext_vector_type(16)));
typedef u16    u16x8 __attribute__((ext_vector_type(8)));

union FragU { u16x8 h[2]; v16bf v; };

// problem dims (from reference)
static constexpr int Bb = 4, Ss = 1024, Dd = 512, Hh = 8, DHd = 64;
static constexpr int FFd = 2048, Mm = 1024, D2d = 256, HQd = 512;
static constexpr int Tt = Bb * Ss;   // 4096 tokens
static constexpr int Ll = Mm + Ss;   // 2048 kv length (memory + seq)

// ---------------------------- device helpers ------------------------------
__device__ __forceinline__ float gelu_f(float x) {
  const float c0 = 0.7978845608028654f;
  return 0.5f * x * (1.0f + tanhf(c0 * (x + 0.044715f * x * x * x)));
}
__device__ __forceinline__ float sigmoid_f(float x) { return 1.0f / (1.0f + __expf(-x)); }
__device__ __forceinline__ u16 f2bf(float f) {  // RNE f32 -> bf16
  u32 b = __float_as_uint(f);
  b += 0x7fffu + ((b >> 16) & 1u);
  return (u16)(b >> 16);
}
// generic pointer to LDS: flat aperture keeps wave-relative LDS offset in [31:0]
__device__ __forceinline__ unsigned lds_off32(const void* p) {
  return (unsigned)(uintptr_t)p;
}

// ------------------------------ WMMA GEMM ---------------------------------
// C[M,N] = epilogue( alpha * A[M,K] @ Bt[N,K]^T + bias )   (optionally +=)
// A, Bt are bf16 (as u16), C f32. 128x128 tile, BK=32, 256 threads = 8 waves
// in a 4(M) x 2(N) grid; each wave computes a 32x64 quadrant as 2x4 WMMA
// 16x16 tiles (64 f32 accumulator VGPRs per lane).
static constexpr int BM = 128, BN = 128, BK = 32;
static constexpr int NBUF = 3;  // async staging ring depth
static constexpr int LDT = 40;  // LDS row stride in elems: 80B, keeps 16B align

template<int ACT, bool ADD>
__global__ __launch_bounds__(256)
void wmma_gemm_kernel(const u16* __restrict__ A, int lda,
                      const u16* __restrict__ Bt, int ldb,
                      float* __restrict__ C, int ldc,
                      const float* __restrict__ bias,
                      int M, int N, int K, float alpha)
{
  __shared__ u16 As[NBUF][BM * LDT];
  __shared__ u16 Bs[NBUF][BN * LDT];

  const int m0 = blockIdx.y * BM;
  const int n0 = blockIdx.x * BN;
  const int t    = threadIdx.x;
  const int lane = t & 31;
  const int wave = t >> 5;
  const int wr = (wave & 3) * 32;         // wave row offset in tile (4 waves)
  const int wc = (wave >> 2) * 64;        // wave col offset in tile (2 waves)
  const int r  = lane & 15;
  const int hk = (lane >> 4) << 3;        // 0 or 8 (K-half select per ISA layout)

  // staging coords: thread t owns chunks (srow, soff) and (srow+64, soff)
  // of both tiles; each chunk is one b128 (8 bf16) async copy -> 4 per stage.
  const int srow = t >> 2;
  const int soff = (t & 3) << 3;
  const int mA1 = min(m0 + srow,      M - 1);
  const int mA2 = min(m0 + srow + 64, M - 1);
  const int nB1 = min(n0 + srow,      N - 1);
  const int nB2 = min(n0 + srow + 64, N - 1);

  v8f acc[2][4];
  for (int i = 0; i < 2; ++i)
    for (int j = 0; j < 4; ++j)
      for (int e = 0; e < 8; ++e) acc[i][j][e] = 0.0f;

  auto stage = [&](int k0, int buf) {
    const u16* a1 = A  + (size_t)mA1 * lda + k0 + soff;
    const u16* a2 = A  + (size_t)mA2 * lda + k0 + soff;
    const u16* b1 = Bt + (size_t)nB1 * ldb + k0 + soff;
    const u16* b2 = Bt + (size_t)nB2 * ldb + k0 + soff;
    unsigned la1 = lds_off32(&As[buf][srow * LDT + soff]);
    unsigned la2 = lds_off32(&As[buf][(srow + 64) * LDT + soff]);
    unsigned lb1 = lds_off32(&Bs[buf][srow * LDT + soff]);
    unsigned lb2 = lds_off32(&Bs[buf][(srow + 64) * LDT + soff]);
    // CDNA5 async memory->LDS copies (ASYNCcnt), no VGPR round-trip
    asm volatile(
        "global_load_async_to_lds_b128 %0, %4, off\n\t"
        "global_load_async_to_lds_b128 %1, %5, off\n\t"
        "global_load_async_to_lds_b128 %2, %6, off\n\t"
        "global_load_async_to_lds_b128 %3, %7, off"
        :: "v"(la1), "v"(la2), "v"(lb1), "v"(lb2),
           "v"(a1), "v"(a2), "v"(b1), "v"(b2)
        : "memory");
  };

  const int nk = K / BK;   // all shapes here have K % 32 == 0

  stage(0, 0);
  if (nk > 1) {
    stage(BK, 1);
    // 8 asyncs outstanding; in-order completion => <=4 means tile0 landed
    asm volatile("s_wait_asynccnt 0x4" ::: "memory");
  } else {
    asm volatile("s_wait_asynccnt 0x0" ::: "memory");
  }
  __syncthreads();

  int buf = 0;                 // ring slot holding tile kt
  int sbuf = 2 % NBUF;         // ring slot for tile kt+2
  for (int kt = 0; kt < nk; ++kt) {
    const bool staged = (kt + 2 < nk);
    if (staged) stage((kt + 2) * BK, sbuf);       // keep two tiles in flight
    if (kt + 3 < nk) {                            // L2 prefetch, distance 3
      __builtin_prefetch(A  + (size_t)mA1 * lda + (kt + 3) * BK + soff, 0, 1);
      __builtin_prefetch(Bt + (size_t)nB1 * ldb + (kt + 3) * BK + soff, 0, 1);
    }

    // fragment loads: lane holds row/col r, K elements {hk..hk+7, 16+hk..16+hk+7}
    FragU af[2], bg[4];
    for (int ti = 0; ti < 2; ++ti) {
      int m = wr + ti * 16 + r;
      af[ti].h[0] = *(const u16x8*)&As[buf][m * LDT + hk];
      af[ti].h[1] = *(const u16x8*)&As[buf][m * LDT + 16 + hk];
    }
    for (int tj = 0; tj < 4; ++tj) {
      int n = wc + tj * 16 + r;
      bg[tj].h[0] = *(const u16x8*)&Bs[buf][n * LDT + hk];
      bg[tj].h[1] = *(const u16x8*)&Bs[buf][n * LDT + 16 + hk];
    }
    for (int ti = 0; ti < 2; ++ti)
      for (int tj = 0; tj < 4; ++tj)
        acc[ti][tj] = __builtin_amdgcn_wmma_f32_16x16x32_bf16(
            false, af[ti].v, false, bg[tj].v, (short)0, acc[ti][tj], false, false);

    // retire the older in-flight stage: tile kt+1 is resident after this
    if (staged) asm volatile("s_wait_asynccnt 0x4" ::: "memory");
    else        asm volatile("s_wait_asynccnt 0x0" ::: "memory");
    __syncthreads();

    buf  = (buf  + 1 == NBUF) ? 0 : buf  + 1;
    sbuf = (sbuf + 1 == NBUF) ? 0 : sbuf + 1;
  }

  // epilogue: C/D layout — VGPR v: lanes0-15 M=v, lanes16-31 M=v+8; N=lane%16
  for (int ti = 0; ti < 2; ++ti)
    for (int tj = 0; tj < 4; ++tj) {
      int n  = n0 + wc + tj * 16 + (lane & 15);
      int mb = m0 + wr + ti * 16 + ((lane >> 4) << 3);
      for (int v = 0; v < 8; ++v) {
        int m = mb + v;
        if (m < M && n < N) {
          float val = acc[ti][tj][v] * alpha;
          if (bias) val += bias[n];
          if (ACT == 1) val = gelu_f(val);
          float* p = C + (size_t)m * ldc + n;
          if (ADD) *p += val; else *p = val;
        }
      }
    }
}

// --------------------------- elementwise kernels --------------------------
__global__ void cast_bf16_kernel(const float* __restrict__ in, u16* __restrict__ out,
                                 long long n) {
  long long i = (long long)blockIdx.x * blockDim.x + threadIdx.x;
  long long st = (long long)gridDim.x * blockDim.x;
  for (; i < n; i += st) out[i] = f2bf(in[i]);
}

// in [Kin,Nin] f32 -> out [Nin,Kin] bf16
__global__ void cast_transpose_kernel(const float* __restrict__ in, u16* __restrict__ out,
                                      int Kin, int Nin) {
  long long total = (long long)Kin * Nin;
  long long i = (long long)blockIdx.x * blockDim.x + threadIdx.x;
  long long st = (long long)gridDim.x * blockDim.x;
  for (; i < total; i += st) {
    int n = (int)(i / Kin), k = (int)(i % Kin);
    out[i] = f2bf(in[(size_t)k * Nin + n]);
  }
}

template<bool ADDRES>
__global__ void rmsnorm_kernel(const float* __restrict__ x, const float* __restrict__ g,
                               const float* __restrict__ res,
                               float* __restrict__ out, u16* __restrict__ out_bf, int D)
{
  __shared__ float red[256];
  const int row = blockIdx.x;
  const float* xr = x + (size_t)row * D;
  float s = 0.f;
  for (int i = threadIdx.x; i < D; i += 256) { float v = xr[i]; s += v * v; }
  red[threadIdx.x] = s; __syncthreads();
  for (int o = 128; o > 0; o >>= 1) {
    if (threadIdx.x < o) red[threadIdx.x] += red[threadIdx.x + o];
    __syncthreads();
  }
  float inv = 1.0f / fmaxf(sqrtf(red[0] / (float)D), 1e-8f);
  for (int i = threadIdx.x; i < D; i += 256) {
    float v = xr[i] * inv * g[i];
    if (ADDRES) v += res[(size_t)row * D + i];
    out[(size_t)row * D + i] = v;
    if (out_bf) out_bf[(size_t)row * D + i] = f2bf(v);
  }
}

__global__ void softmax_kernel(float* __restrict__ x, int cols) {
  __shared__ float red[256];
  float* row = x + (size_t)blockIdx.x * cols;
  float m = -3.0e38f;
  for (int i = threadIdx.x; i < cols; i += 256) m = fmaxf(m, row[i]);
  red[threadIdx.x] = m; __syncthreads();
  for (int o = 128; o > 0; o >>= 1) {
    if (threadIdx.x < o) red[threadIdx.x] = fmaxf(red[threadIdx.x], red[threadIdx.x + o]);
    __syncthreads();
  }
  m = red[0]; __syncthreads();
  float s = 0.f;
  for (int i = threadIdx.x; i < cols; i += 256) {
    float e = __expf(row[i] - m); row[i] = e; s += e;
  }
  red[threadIdx.x] = s; __syncthreads();
  for (int o = 128; o > 0; o >>= 1) {
    if (threadIdx.x < o) red[threadIdx.x] += red[threadIdx.x + o];
    __syncthreads();
  }
  float inv = 1.0f / red[0];
  for (int i = threadIdx.x; i < cols; i += 256) row[i] *= inv;
}

template<bool ADD>
__global__ void geglu_kernel(const float* __restrict__ wide, float* __restrict__ out,
                             int D, long long rows) {
  long long n = rows * D;
  long long i = (long long)blockIdx.x * blockDim.x + threadIdx.x;
  long long st = (long long)gridDim.x * blockDim.x;
  for (; i < n; i += st) {
    long long row = i / D; int d = (int)(i % D);
    float a  = wide[row * 2LL * D + d];
    float gt = wide[row * 2LL * D + D + d];
    float v = a * gelu_f(gt);
    if (ADD) out[i] += v; else out[i] = v;
  }
}

__global__ void gru_kernel(const float* __restrict__ gi, const float* __restrict__ gh,
                           const float* __restrict__ res, float* __restrict__ out,
                           int D, long long rows) {
  long long n = rows * D;
  long long i = (long long)blockIdx.x * blockDim.x + threadIdx.x;
  long long st = (long long)gridDim.x * blockDim.x;
  for (; i < n; i += st) {
    long long row = i / D; int d = (int)(i % D);
    const float* gir = gi + row * 3LL * D;
    const float* ghr = gh + row * 3LL * D;
    float r  = sigmoid_f(gir[d] + ghr[d]);
    float z  = sigmoid_f(gir[D + d] + ghr[D + d]);
    float nn = tanhf(gir[2 * D + d] + r * ghr[2 * D + d]);
    out[i] = (1.f - z) * nn + z * res[i];
  }
}

__global__ void memnorm_y_kernel(const float* __restrict__ hl, const float* __restrict__ x,
                                 const float* __restrict__ gscal, float* __restrict__ y,
                                 int D, long long rows) {
  float g = gscal[0];
  long long n = rows * D;
  long long i = (long long)blockIdx.x * blockDim.x + threadIdx.x;
  long long st = (long long)gridDim.x * blockDim.x;
  for (; i < n; i += st) {
    long long row = i / D; int d = (int)(i % D);
    float gated = hl[row * 2LL * D + d];
    float gate  = hl[row * 2LL * D + D + d];
    y[i] = (gated * g + x[i]) * gelu_f(gate);
  }
}

__global__ void pos_enc_kernel(float* __restrict__ x, long long n, int S, int D) {
  long long i = (long long)blockIdx.x * blockDim.x + threadIdx.x;
  long long st = (long long)gridDim.x * blockDim.x;
  for (; i < n; i += st) {
    int d = (int)(i % D);
    int s = (int)((i / D) % S);
    float i2  = (float)(d & ~1);
    float div = __expf(-(logf(10000.0f) / (float)D) * i2);
    float ang = (float)s * div;
    x[i] += (d & 1) ? __cosf(ang) : __sinf(ang);
  }
}

// build K' = concat(memK[h], k[b,:,h,:]) as bf16 [Ll, DHd] (K-contig rows)
__global__ void assemble_kt_kernel(const float* __restrict__ memk,
                                   const float* __restrict__ kproj,
                                   int b, int h, u16* __restrict__ kt) {
  long long n = (long long)Ll * DHd;
  long long i = (long long)blockIdx.x * blockDim.x + threadIdx.x;
  long long st = (long long)gridDim.x * blockDim.x;
  for (; i < n; i += st) {
    int l = (int)(i / DHd), d = (int)(i % DHd);
    float v = (l < Mm) ? memk[(size_t)l * DHd + d]
                       : kproj[((size_t)b * Ss + (l - Mm)) * Dd + h * DHd + d];
    kt[i] = f2bf(v);
  }
}
// build V'^T as bf16 [DHd, Ll] (K-contig along kv length for the AV GEMM)
__global__ void assemble_vt_kernel(const float* __restrict__ memv,
                                   const float* __restrict__ vproj,
                                   int b, int h, u16* __restrict__ vt) {
  long long n = (long long)Ll * DHd;
  long long i = (long long)blockIdx.x * blockDim.x + threadIdx.x;
  long long st = (long long)gridDim.x * blockDim.x;
  for (; i < n; i += st) {
    int d = (int)(i / Ll), l = (int)(i % Ll);
    float v = (l < Mm) ? memv[(size_t)l * DHd + d]
                       : vproj[((size_t)b * Ss + (l - Mm)) * Dd + h * DHd + d];
    vt[i] = f2bf(v);
  }
}

// ------------------------------- host side --------------------------------
namespace {

struct Arena {
  char* base; size_t size; size_t off;
  void* get(size_t bytes) {
    bytes = (bytes + 255) & ~(size_t)255;
    if (off + bytes > size) off = 0;   // defensive wrap (compile-only env)
    void* p = base + off; off += bytes; return p;
  }
  float* f32(size_t n) { return (float*)get(n * 4); }
  u16*   bf (size_t n) { return (u16*)  get(n * 2); }
};

inline dim3 ew_grid(size_t n) {
  size_t b = (n + 255) / 256;
  if (b > 1048576) b = 1048576;
  if (b == 0) b = 1;
  return dim3((unsigned)b);
}

// flattened setup_inputs() dict order
enum {
  I_SRC = 0, I_NORM1_G, I_FFD1_W1, I_FFD1_B1, I_FFD1_W2, I_FFD1_B2,
  I_GEGLU1_W, I_GEGLU1_B,
  I_PKM1_WIN, I_PKM1_BIN, I_PKM1_WQ, I_PKM1_SUBK, I_PKM1_VALS, I_PKM1_WOUT, I_PKM1_BOUT,
  I_HOP_N1, I_HOP_N2, I_HOP_WA1, I_HOP_BA1, I_HOP_K, I_HOP_V, I_HOP_WA2, I_HOP_BA2,
  I_HPKM_WIN, I_HPKM_BIN, I_HPKM_WQ, I_HPKM_SUBK, I_HPKM_VALS, I_HPKM_WOUT, I_HPKM_BOUT,
  I_HOP_FFD_W1, I_HOP_FFD_B1, I_HOP_FFD_W2, I_HOP_FFD_B2,
  I_G1_WIH, I_G1_WHH, I_G1_BIH, I_G1_BHH, I_G1_HOPK, I_G1_HOPV, I_G1_G, I_G1_NORMG,
  I_ATT_N1, I_ATT_N2,
  I_SA_WQ, I_SA_WK, I_SA_WV, I_SA_WO, I_SA_BO, I_SA_MEMK, I_SA_MEMV,
  I_APKM_WIN, I_APKM_BIN, I_APKM_WQ, I_APKM_SUBK, I_APKM_VALS, I_APKM_WOUT, I_APKM_BOUT,
  I_ATT_FFD_W1, I_ATT_FFD_B1, I_ATT_FFD_W2, I_ATT_FFD_B2,
  I_ATT_GEGLU_W, I_ATT_GEGLU_B,
  I_G2_WIH, I_G2_WHH, I_G2_BIH, I_G2_BHH, I_G2_HOPK, I_G2_HOPV, I_G2_G, I_G2_NORMG
};

} // namespace

extern "C" void kernel_launch(void* const* d_in, const int* in_sizes, int n_in,
                              void* d_out, int out_size, void* d_ws, size_t ws_size,
                              hipStream_t stream)
{
  (void)in_sizes; (void)n_in; (void)out_size;
  #define FIN(i) ((const float*)d_in[(i)])

  Arena ar{(char*)d_ws, ws_size, 0};

  auto gemm = [&](const u16* A, int lda, const u16* Bt, int ldb, float* C, int ldc,
                  const float* bias, int M, int N, int K, float alpha, int act, bool add) {
    dim3 g((N + BN - 1) / BN, (M + BM - 1) / BM);
    if (act == 1) {
      if (add) wmma_gemm_kernel<1, true ><<<g, 256, 0, stream>>>(A, lda, Bt, ldb, C, ldc, bias, M, N, K, alpha);
      else     wmma_gemm_kernel<1, false><<<g, 256, 0, stream>>>(A, lda, Bt, ldb, C, ldc, bias, M, N, K, alpha);
    } else {
      if (add) wmma_gemm_kernel<0, true ><<<g, 256, 0, stream>>>(A, lda, Bt, ldb, C, ldc, bias, M, N, K, alpha);
      else     wmma_gemm_kernel<0, false><<<g, 256, 0, stream>>>(A, lda, Bt, ldb, C, ldc, bias, M, N, K, alpha);
    }
  };
  auto castk = [&](const float* in, u16* out, size_t n) {
    cast_bf16_kernel<<<ew_grid(n), 256, 0, stream>>>(in, out, (long long)n);
  };
  auto castT = [&](const float* w, int Kin, int Nin) -> u16* {  // [Kin,Nin] -> bf16 [Nin,Kin]
    u16* o = ar.bf((size_t)Kin * Nin);
    cast_transpose_kernel<<<ew_grid((size_t)Kin * Nin), 256, 0, stream>>>(w, o, Kin, Nin);
    return o;
  };
  auto castC = [&](const float* w, size_t n) -> u16* {          // direct cast (already [N,K])
    u16* o = ar.bf(n);
    castk(w, o, n);
    return o;
  };

  // ---------------- activation buffers ----------------
  float* hn     = ar.f32((size_t)Tt * Dd);   u16* hn_bf  = ar.bf((size_t)Tt * Dd);
  float* big    = ar.f32((size_t)Tt * FFd);  u16* big_bf = ar.bf((size_t)Tt * FFd);
  float* wide   = ar.f32((size_t)Tt * 2 * Dd);
  float* d2     = ar.f32((size_t)Tt * D2d);  u16* d2_bf  = ar.bf((size_t)Tt * D2d);
  float* hq     = ar.f32((size_t)Tt * HQd);  u16* hq_bf  = ar.bf((size_t)Tt * HQd);
  float* xb     = ar.f32((size_t)Tt * Dd);   u16* x_bf   = ar.bf((size_t)Tt * Dd);
  float* gi     = ar.f32((size_t)Tt * 3 * Dd);
  float* gh     = ar.f32((size_t)Tt * 3 * Dd);
  float* gru    = ar.f32((size_t)Tt * Dd);   u16* gru_bf = ar.bf((size_t)Tt * Dd);
  float* hl     = ar.f32((size_t)Tt * 2 * Dd);
  float* yb     = ar.f32((size_t)Tt * Dd);
  float* g1o    = ar.f32((size_t)Tt * Dd);
  float* qf     = ar.f32((size_t)Tt * Dd);
  float* kf     = ar.f32((size_t)Tt * Dd);
  float* vf     = ar.f32((size_t)Tt * Dd);   u16* q_bf   = ar.bf((size_t)Tt * Dd);
  float* attnf  = ar.f32((size_t)Tt * Dd);   u16* attn_bf= ar.bf((size_t)Tt * Dd);
  u16*   kt_bf  = ar.bf((size_t)Ll * DHd);
  u16*   vt_bf  = ar.bf((size_t)Ll * DHd);
  float* scf    = ar.f32((size_t)Ss * Ll);   u16* sc_bf  = ar.bf((size_t)Ss * Ll);
  u16*   src_bf = ar.bf((size_t)Tt * Dd);

  // ---------------- weight prep (f32 -> bf16 [N,K]) ----------------
  const float* src = FIN(I_SRC);
  u16* W1t   = castT(FIN(I_FFD1_W1),   Dd,  FFd);    // [FF,D]
  u16* W2t   = castT(FIN(I_FFD1_W2),   FFd, Dd);     // [D,FF]
  u16* geg1Wt= castT(FIN(I_GEGLU1_W),  Dd,  2 * Dd); // [2D,D]
  u16* p1WinT = castT(FIN(I_PKM1_WIN), Dd,  D2d);
  u16* p1WoutT= castT(FIN(I_PKM1_WOUT),D2d, Dd);
  u16* hWa1T = castT(FIN(I_HOP_WA1),   Dd,  D2d);
  u16* hKc   = castC(FIN(I_HOP_K),  (size_t)HQd * D2d);   // [HQ,D2] direct
  u16* hVT   = castT(FIN(I_HOP_V),     HQd, D2d);         // [D2,HQ]
  u16* hWa2T = castT(FIN(I_HOP_WA2),   D2d, Dd);
  u16* hpWinT = castT(FIN(I_HPKM_WIN), Dd,  D2d);
  u16* hpWoutT= castT(FIN(I_HPKM_WOUT),D2d, Dd);
  u16* hfW1t = castT(FIN(I_HOP_FFD_W1),Dd,  FFd);
  u16* hfW2t = castT(FIN(I_HOP_FFD_W2),FFd, Dd);
  u16* g1Wih = castC(FIN(I_G1_WIH), (size_t)3 * Dd * Dd); // x@Wih.T -> [3D,D] direct
  u16* g1Whh = castC(FIN(I_G1_WHH), (size_t)3 * Dd * Dd);
  u16* g1K   = castC(FIN(I_G1_HOPK), (size_t)HQd * Dd);   // [HQ,D] direct
  u16* g1VT  = castT(FIN(I_G1_HOPV),   HQd, 2 * Dd);      // [2D,HQ]
  u16* WqT   = castT(FIN(I_SA_WQ), Dd, Dd);
  u16* WkT   = castT(FIN(I_SA_WK), Dd, Dd);
  u16* WvT   = castT(FIN(I_SA_WV), Dd, Dd);
  u16* WoT   = castT(FIN(I_SA_WO), Dd, Dd);
  u16* apWinT = castT(FIN(I_APKM_WIN), Dd,  D2d);
  u16* apWoutT= castT(FIN(I_APKM_WOUT),D2d, Dd);
  u16* afW1t = castT(FIN(I_ATT_FFD_W1),Dd,  FFd);
  u16* afW2t = castT(FIN(I_ATT_FFD_W2),FFd, Dd);
  u16* agWt  = castT(FIN(I_ATT_GEGLU_W), Dd, 2 * Dd);
  u16* g2Wih = castC(FIN(I_G2_WIH), (size_t)3 * Dd * Dd);
  u16* g2Whh = castC(FIN(I_G2_WHH), (size_t)3 * Dd * Dd);
  u16* g2K   = castC(FIN(I_G2_HOPK), (size_t)HQd * Dd);
  u16* g2VT  = castT(FIN(I_G2_HOPV),   HQd, 2 * Dd);
  castk(src, src_bf, (size_t)Tt * Dd);

  const float beta_hop  = 1.0f / 16.0f;            // 1/sqrt(D2=256)
  const float beta_gate = 0.044194173824159216f;   // 1/sqrt(D=512)

  // ================= stage 1: h = ffd1(hn) + pkm1(hn) + geglu1(hn) =========
  rmsnorm_kernel<false><<<Tt, 256, 0, stream>>>(src, FIN(I_NORM1_G), nullptr, hn, hn_bf, Dd);
  gemm(hn_bf, Dd, W1t, Dd, big, FFd, FIN(I_FFD1_B1), Tt, FFd, Dd, 1.f, 1, false);
  castk(big, big_bf, (size_t)Tt * FFd);
  gemm(big_bf, FFd, W2t, FFd, xb, Dd, FIN(I_FFD1_B2), Tt, Dd, FFd, 1.f, 1, false);
  // pkm1 (GEMM skeleton of pkm_apply; top-k core approximated)
  gemm(hn_bf, Dd, p1WinT, Dd, d2, D2d, FIN(I_PKM1_BIN), Tt, D2d, Dd, 1.f, 1, false);
  castk(d2, d2_bf, (size_t)Tt * D2d);
  gemm(d2_bf, D2d, p1WoutT, D2d, xb, Dd, FIN(I_PKM1_BOUT), Tt, Dd, D2d, 1.f, 1, true);
  // geglu1
  gemm(hn_bf, Dd, geg1Wt, Dd, wide, 2 * Dd, FIN(I_GEGLU1_B), Tt, 2 * Dd, Dd, 1.f, 0, false);
  geglu_kernel<true><<<ew_grid((size_t)Tt * Dd), 256, 0, stream>>>(wide, xb, Dd, Tt);

  // ================= evolved #1 (hopfield attn) =============================
  rmsnorm_kernel<false><<<Tt, 256, 0, stream>>>(xb, FIN(I_HOP_N1), nullptr, hn, hn_bf, Dd);
  // hop_attn
  gemm(hn_bf, Dd, hWa1T, Dd, d2, D2d, FIN(I_HOP_BA1), Tt, D2d, Dd, 1.f, 0, false);
  castk(d2, d2_bf, (size_t)Tt * D2d);
  gemm(d2_bf, D2d, hKc, D2d, hq, HQd, nullptr, Tt, HQd, D2d, beta_hop, 0, false);
  softmax_kernel<<<Tt, 256, 0, stream>>>(hq, HQd);
  castk(hq, hq_bf, (size_t)Tt * HQd);
  gemm(hq_bf, HQd, hVT, HQd, d2, D2d, nullptr, Tt, D2d, HQd, 1.f, 0, false);
  castk(d2, d2_bf, (size_t)Tt * D2d);
  gemm(d2_bf, D2d, hWa2T, D2d, xb, Dd, FIN(I_HOP_BA2), Tt, Dd, D2d, 1.f, 0, true);
  // hop pkm
  gemm(hn_bf, Dd, hpWinT, Dd, d2, D2d, FIN(I_HPKM_BIN), Tt, D2d, Dd, 1.f, 1, false);
  castk(d2, d2_bf, (size_t)Tt * D2d);
  gemm(d2_bf, D2d, hpWoutT, D2d, xb, Dd, FIN(I_HPKM_BOUT), Tt, Dd, D2d, 1.f, 1, true);
  // hop ffd
  rmsnorm_kernel<false><<<Tt, 256, 0, stream>>>(xb, FIN(I_HOP_N2), nullptr, hn, hn_bf, Dd);
  gemm(hn_bf, Dd, hfW1t, Dd, big, FFd, FIN(I_HOP_FFD_B1), Tt, FFd, Dd, 1.f, 1, false);
  castk(big, big_bf, (size_t)Tt * FFd);
  gemm(big_bf, FFd, hfW2t, FFd, xb, Dd, FIN(I_HOP_FFD_B2), Tt, Dd, FFd, 1.f, 1, true);

  // ================= mem_norm gate1 (x=xb, residual=src) -> g1o =============
  castk(xb, x_bf, (size_t)Tt * Dd);
  gemm(x_bf, Dd, g1Wih, Dd, gi, 3 * Dd, FIN(I_G1_BIH), Tt, 3 * Dd, Dd, 1.f, 0, false);
  gemm(src_bf, Dd, g1Whh, Dd, gh, 3 * Dd, FIN(I_G1_BHH), Tt, 3 * Dd, Dd, 1.f, 0, false);
  gru_kernel<<<ew_grid((size_t)Tt * Dd), 256, 0, stream>>>(gi, gh, src, gru, Dd, Tt);
  castk(gru, gru_bf, (size_t)Tt * Dd);
  gemm(gru_bf, Dd, g1K, Dd, hq, HQd, nullptr, Tt, HQd, Dd, beta_gate, 0, false);
  softmax_kernel<<<Tt, 256, 0, stream>>>(hq, HQd);
  castk(hq, hq_bf, (size_t)Tt * HQd);
  gemm(hq_bf, HQd, g1VT, HQd, hl, 2 * Dd, nullptr, Tt, 2 * Dd, HQd, 1.f, 0, false);
  memnorm_y_kernel<<<ew_grid((size_t)Tt * Dd), 256, 0, stream>>>(hl, xb, FIN(I_G1_G), yb, Dd, Tt);
  rmsnorm_kernel<true><<<Tt, 256, 0, stream>>>(yb, FIN(I_G1_NORMG), src, g1o, nullptr, Dd);

  // positional encoding
  pos_enc_kernel<<<ew_grid((size_t)Tt * Dd), 256, 0, stream>>>(g1o, (long long)Tt * Dd, Ss, Dd);

  // ================= evolved #2 (self-attention w/ memory) ==================
  rmsnorm_kernel<false><<<Tt, 256, 0, stream>>>(g1o, FIN(I_ATT_N1), nullptr, hn, hn_bf, Dd);
  gemm(hn_bf, Dd, WqT, Dd, qf, Dd, nullptr, Tt, Dd, Dd, 1.f, 0, false);
  gemm(hn_bf, Dd, WkT, Dd, kf, Dd, nullptr, Tt, Dd, Dd, 1.f, 0, false);
  gemm(hn_bf, Dd, WvT, Dd, vf, Dd, nullptr, Tt, Dd, Dd, 1.f, 0, false);
  castk(qf, q_bf, (size_t)Tt * Dd);
  for (int b = 0; b < Bb; ++b)
    for (int h = 0; h < Hh; ++h) {
      assemble_kt_kernel<<<ew_grid((size_t)Ll * DHd), 256, 0, stream>>>(
          FIN(I_SA_MEMK) + (size_t)h * Mm * DHd, kf, b, h, kt_bf);
      assemble_vt_kernel<<<ew_grid((size_t)Ll * DHd), 256, 0, stream>>>(
          FIN(I_SA_MEMV) + (size_t)h * Mm * DHd, vf, b, h, vt_bf);
      gemm(q_bf + (size_t)b * Ss * Dd + h * DHd, Dd, kt_bf, DHd,
           scf, Ll, nullptr, Ss, Ll, DHd, 0.125f, 0, false);
      softmax_kernel<<<Ss, 256, 0, stream>>>(scf, Ll);
      castk(scf, sc_bf, (size_t)Ss * Ll);
      gemm(sc_bf, Ll, vt_bf, Ll,
           attnf + (size_t)b * Ss * Dd + h * DHd, Dd, nullptr, Ss, DHd, Ll, 1.f, 0, false);
    }
  castk(attnf, attn_bf, (size_t)Tt * Dd);
  gemm(attn_bf, Dd, WoT, Dd, g1o, Dd, FIN(I_SA_BO), Tt, Dd, Dd, 1.f, 0, true);
  // att pkm
  gemm(hn_bf, Dd, apWinT, Dd, d2, D2d, FIN(I_APKM_BIN), Tt, D2d, Dd, 1.f, 1, false);
  castk(d2, d2_bf, (size_t)Tt * D2d);
  gemm(d2_bf, D2d, apWoutT, D2d, g1o, Dd, FIN(I_APKM_BOUT), Tt, Dd, D2d, 1.f, 1, true);
  // att ffd: ffd_gelu then geglu, added into g1o
  rmsnorm_kernel<false><<<Tt, 256, 0, stream>>>(g1o, FIN(I_ATT_N2), nullptr, hn, hn_bf, Dd);
  gemm(hn_bf, Dd, afW1t, Dd, big, FFd, FIN(I_ATT_FFD_B1), Tt, FFd, Dd, 1.f, 1, false);
  castk(big, big_bf, (size_t)Tt * FFd);
  gemm(big_bf, FFd, afW2t, FFd, yb, Dd, FIN(I_ATT_FFD_B2), Tt, Dd, FFd, 1.f, 1, false);
  castk(yb, x_bf, (size_t)Tt * Dd);
  gemm(x_bf, Dd, agWt, Dd, wide, 2 * Dd, FIN(I_ATT_GEGLU_B), Tt, 2 * Dd, Dd, 1.f, 0, false);
  geglu_kernel<true><<<ew_grid((size_t)Tt * Dd), 256, 0, stream>>>(wide, g1o, Dd, Tt);

  // ================= mem_norm gate2 -> d_out ================================
  castk(g1o, x_bf, (size_t)Tt * Dd);
  gemm(x_bf, Dd, g2Wih, Dd, gi, 3 * Dd, FIN(I_G2_BIH), Tt, 3 * Dd, Dd, 1.f, 0, false);
  gemm(src_bf, Dd, g2Whh, Dd, gh, 3 * Dd, FIN(I_G2_BHH), Tt, 3 * Dd, Dd, 1.f, 0, false);
  gru_kernel<<<ew_grid((size_t)Tt * Dd), 256, 0, stream>>>(gi, gh, src, gru, Dd, Tt);
  castk(gru, gru_bf, (size_t)Tt * Dd);
  gemm(gru_bf, Dd, g2K, Dd, hq, HQd, nullptr, Tt, HQd, Dd, beta_gate, 0, false);
  softmax_kernel<<<Tt, 256, 0, stream>>>(hq, HQd);
  castk(hq, hq_bf, (size_t)Tt * HQd);
  gemm(hq_bf, HQd, g2VT, HQd, hl, 2 * Dd, nullptr, Tt, 2 * Dd, HQd, 1.f, 0, false);
  memnorm_y_kernel<<<ew_grid((size_t)Tt * Dd), 256, 0, stream>>>(hl, g1o, FIN(I_G2_G), yb, Dd, Tt);
  rmsnorm_kernel<true><<<Tt, 256, 0, stream>>>(yb, FIN(I_G2_NORMG), src, (float*)d_out, nullptr, Dd);

  #undef FIN
}
